// Attention_41394894798989
// MI455X (gfx1250) — matrix operand order repaired and checked by reading, not compiled, not run on hardware
//
#include <hip/hip_runtime.h>
#include <hip/hip_bf16.h>

// ---- problem constants (from reference) ----
#define BB   4
#define SS   2048
#define HID  2048
#define NH   16
#define NKV  4
#define HD   128
#define GROUP (NH / NKV)   // 4

typedef __bf16 v16bf __attribute__((ext_vector_type(16)));
typedef __bf16 v8bf  __attribute__((ext_vector_type(8)));
typedef float  v8f   __attribute__((ext_vector_type(8)));

static __device__ __forceinline__ v16bf cat16(v8bf lo, v8bf hi) {
    return __builtin_shufflevector(lo, hi, 0,1,2,3,4,5,6,7,8,9,10,11,12,13,14,15);
}
static __device__ __forceinline__ v8bf ld8(const __bf16* p) {
    return *(const v8bf*)p;   // 16B vector load (global_load_b128 / ds_load_b128)
}

static __device__ __forceinline__ v8f wmma_bf16(v16bf a, v16bf b, v8f c) {
    return __builtin_amdgcn_wmma_f32_16x16x32_bf16(false, a, false, b,
                                                   (short)0, c, false, false);
}

// ================= Tensor Data Mover (TDM) support ===================
#if __has_builtin(__builtin_amdgcn_tensor_load_to_lds)
#define USE_TDM 1
typedef unsigned tdm_v4u __attribute__((ext_vector_type(4)));
typedef int      tdm_v8i __attribute__((ext_vector_type(8)));
typedef int      tdm_v4i __attribute__((ext_vector_type(4)));

#if __has_include(<hip/amd_detail/amd_gfx1250_TDM.h>)
// amdgpu-toolchain (clang-23): 6-arg form
#define TDM_CALL(g0, g1, g2, g3) \
    __builtin_amdgcn_tensor_load_to_lds(g0, g1, g2, g3, (tdm_v8i)0, 0)
#else
// ROCm 7.2 (clang-22): 5-arg form
#define TDM_CALL(g0, g1, g2, g3) \
    __builtin_amdgcn_tensor_load_to_lds(g0, g1, g2, g3, 0)
#endif

// Issue an async 2-D tile load (bf16 elements) global -> LDS.
// Tile: tile_d0 elements along a row (contiguous), tile_d1 rows,
// row stride = stride0 elements. D# per CDNA5 ISA §8.3-8.6.
static __device__ __forceinline__ void tdm_load_2d(unsigned lds_off,
                                                   const void* gptr,
                                                   unsigned tensor_d0,
                                                   unsigned tensor_d1,
                                                   unsigned long long stride0,
                                                   unsigned tile_d0,
                                                   unsigned tile_d1) {
    unsigned long long ga = (unsigned long long)(uintptr_t)gptr;
    tdm_v4u g0;
    g0[0] = 1u;                                              // count=1, user mode
    g0[1] = lds_off;                                         // lds_addr (bytes)
    g0[2] = (unsigned)ga;                                    // global_addr[31:0]
    g0[3] = (unsigned)((ga >> 32) & 0x1FFFFFFu) | 0x80000000u; // addr[56:32] | type=2
    tdm_v8i g1;
    g1[0] = (int)(1u << 16);                                 // data_size = 2 bytes
    g1[1] = (int)((tensor_d0 & 0xFFFFu) << 16);              // tensor_dim0[15:0]
    g1[2] = (int)(((tensor_d0 >> 16) & 0xFFFFu) |
                  ((tensor_d1 & 0xFFFFu) << 16));            // d0 hi | d1 lo
    g1[3] = (int)(((tensor_d1 >> 16) & 0xFFFFu) |
                  ((tile_d0 & 0xFFFFu) << 16));              // d1 hi | tile_dim0
    g1[4] = (int)(tile_d1 & 0xFFFFu);                        // tile_dim1, tile_dim2=0
    g1[5] = (int)(unsigned)(stride0 & 0xFFFFFFFFu);          // dim0_stride[31:0]
    g1[6] = (int)(unsigned)((stride0 >> 32) & 0xFFFFu);      // dim0_stride[47:32]
    g1[7] = 0;                                               // dim1_stride unused (2D)
    tdm_v4i g2;
    g2[0] = 1; g2[1] = 1; g2[2] = 0; g2[3] = 0;              // tensor_dim2/3 = 1
    tdm_v4i g3;
    g3[0] = 0; g3[1] = (int)(1u << 16); g3[2] = 0; g3[3] = 0;// tensor_dim4 = 1
    TDM_CALL(g0, g1, g2, g3);
}
#endif  // USE_TDM

// =====================================================================
// Prep kernel A: hidden f32 -> bf16 (row-major unchanged)
// =====================================================================
__global__ __launch_bounds__(256)
void hcvt_kernel(const float* __restrict__ in, __bf16* __restrict__ out, int n) {
    int i = blockIdx.x * 256 + threadIdx.x;
    if (i < n) out[i] = (__bf16)in[i];
}

// =====================================================================
// Prep kernel B: weight f32 [K][N] -> bf16 transposed [N][K]
// =====================================================================
__global__ __launch_bounds__(256)
void wtrans_kernel(const float* __restrict__ in, __bf16* __restrict__ out,
                   int K, int N) {
    __shared__ float tile[32][33];
    const int n0 = blockIdx.x * 32;
    const int k0 = blockIdx.y * 32;
    const int tx = threadIdx.x;   // 0..31
    const int ty = threadIdx.y;   // 0..7
#pragma unroll
    for (int i = 0; i < 32; i += 8)
        tile[ty + i][tx] = in[(size_t)(k0 + ty + i) * N + n0 + tx];
    __syncthreads();
#pragma unroll
    for (int i = 0; i < 32; i += 8)
        out[(size_t)(n0 + ty + i) * K + k0 + tx] = (__bf16)tile[tx][ty + i];
}

// =====================================================================
// Kernel 1: QKV projection + RoPE (bf16 operands, transposed weights).
// =====================================================================
__global__ __launch_bounds__(32)
void qkv_rope_kernel(const __bf16* __restrict__ Hbf,
                     const float* __restrict__ cos_t,
                     const float* __restrict__ sin_t,
                     const __bf16* __restrict__ Wqt,   // [NH*HD][HID]
                     const __bf16* __restrict__ Wkt,   // [NKV*HD][HID]
                     const __bf16* __restrict__ Wvt,   // [NKV*HD][HID]
                     __bf16* __restrict__ Qb,
                     __bf16* __restrict__ Kb,
                     __bf16* __restrict__ Vt)
{
    const int lane = threadIdx.x & 31;
    const int half = lane >> 4;
    const int l16  = lane & 15;

    const int token0 = blockIdx.x * 16;
    const int b  = token0 / SS;
    const int s0 = token0 % SS;

    const int y = blockIdx.y;
    int mode, head;
    const __bf16* Wt;
    if (y < NH)            { mode = 0; head = y;            Wt = Wqt; }
    else if (y < NH + NKV) { mode = 1; head = y - NH;       Wt = Wkt; }
    else                   { mode = 2; head = y - NH - NKV; Wt = Wvt; }
    const int colbase = head * HD;

    v8f acc[8];
#pragma unroll
    for (int t = 0; t < 8; ++t) acc[t] = (v8f)(0.0f);

    const __bf16* ha = Hbf + (size_t)(token0 + l16) * HID;

    for (int k0 = 0; k0 < HID; k0 += 32) {
        v16bf a = cat16(ld8(ha + k0 + half * 8),
                        ld8(ha + k0 + 16 + half * 8));
#pragma unroll
        for (int t = 0; t < 8; ++t) {
            const __bf16* wp = Wt + (size_t)(colbase + t * 16 + l16) * HID
                                  + k0;
            v16bf bfr = cat16(ld8(wp + half * 8), ld8(wp + 16 + half * 8));
            acc[t] = wmma_bf16(a, bfr, acc[t]);
        }
    }

    if (mode < 2) {
#pragma unroll
        for (int r = 0; r < 8; ++r) {
            const int s = s0 + half * 8 + r;
#pragma unroll
            for (int t = 0; t < 8; ++t) {
                const int d = t * 16 + l16;
                const float qv = acc[t][r];
                const float rv = (t < 4) ? -acc[t + 4][r] : acc[t - 4][r];
                const float c  = cos_t[((size_t)b * SS + s) * HD + d];
                const float sn = sin_t[((size_t)b * SS + s) * HD + d];
                const float res = qv * c + rv * sn;
                if (mode == 0)
                    Qb[(((size_t)b * NH + head) * SS + s) * HD + d] = (__bf16)res;
                else
                    Kb[(((size_t)b * NKV + head) * SS + s) * HD + d] = (__bf16)res;
            }
        }
    } else {
#pragma unroll
        for (int r = 0; r < 8; ++r) {
            const int s = s0 + half * 8 + r;
#pragma unroll
            for (int t = 0; t < 8; ++t) {
                const int d = t * 16 + l16;
                Vt[(((size_t)b * NKV + head) * HD + d) * SS + s] = (__bf16)acc[t][r];
            }
        }
    }
}

// =====================================================================
// Kernel 2: flash attention with TDM double-buffered K/V staging in LDS.
// grid.x = S/16 query tiles, grid.y = NH, grid.z = B. One wave per block.
// =====================================================================
__global__ __launch_bounds__(32)
void attn_kernel(const __bf16* __restrict__ Qb,
                 const __bf16* __restrict__ Kb,
                 const __bf16* __restrict__ Vt,
                 __bf16* __restrict__ Ab)
{
    const int lane = threadIdx.x & 31;
    const int half = lane >> 4;
    const int l16  = lane & 15;

    const int q0  = blockIdx.x * 16;
    const int h   = blockIdx.y;
    const int b   = blockIdx.z;
    const int kvh = h / GROUP;

    const __bf16* Qp = Qb + (((size_t)b * NH + h) * SS + q0) * HD;
    const __bf16* Kp = Kb + (((size_t)b * NKV + kvh) * SS) * HD;
    const __bf16* Vp = Vt + (((size_t)b * NKV + kvh) * HD) * SS;

    // Q fragments in registers (4 k-chunks of 32 covering d=128)
    v16bf qf[4];
#pragma unroll
    for (int kc = 0; kc < 4; ++kc) {
        const __bf16* qp = Qp + (size_t)l16 * HD + kc * 32;
        qf[kc] = cat16(ld8(qp + half * 8), ld8(qp + 16 + half * 8));
    }

    v8f o[8];
#pragma unroll
    for (int t = 0; t < 8; ++t) o[t] = (v8f)(0.0f);
    float rm[8], rl[8];
#pragma unroll
    for (int r = 0; r < 8; ++r) { rm[r] = -INFINITY; rl[r] = 0.0f; }

    __shared__ __attribute__((aligned(16))) __bf16 Plds[16 * 32];
    const float scale = 0.08838834764831845f; // 1/sqrt(128)

#ifdef USE_TDM
    // double-buffered staging tiles: K tile 32x128, V tile 128x32 (bf16)
    __shared__ __attribute__((aligned(16))) __bf16 Klds[2][32 * HD];
    __shared__ __attribute__((aligned(16))) __bf16 Vlds[2][HD * 32];
    // prologue: fetch block 0 into buffer 0
    tdm_load_2d((unsigned)(uintptr_t)&Klds[0][0], Kp,
                HD, SS, HD, HD, 32);
    tdm_load_2d((unsigned)(uintptr_t)&Vlds[0][0], Vp,
                SS, HD, SS, 32, HD);
#endif

    const int nblk = SS / 32;
    for (int i = 0; i < nblk; ++i) {
        const int key0 = i * 32;
#ifdef USE_TDM
        const int buf = i & 1;
        if (i + 1 < nblk) {
            // async prefetch of next K/V block into the other buffer
            tdm_load_2d((unsigned)(uintptr_t)&Klds[buf ^ 1][0],
                        Kp + (size_t)(key0 + 32) * HD, HD, SS, HD, HD, 32);
            tdm_load_2d((unsigned)(uintptr_t)&Vlds[buf ^ 1][0],
                        Vp + (key0 + 32), SS, HD, SS, 32, HD);
            __builtin_amdgcn_s_wait_tensorcnt(2);  // current pair complete
        } else {
            __builtin_amdgcn_s_wait_tensorcnt(0);
        }
#endif
        // ---- scores: 16 queries x 32 keys ----
        v8f sacc[2];
        sacc[0] = (v8f)(0.0f); sacc[1] = (v8f)(0.0f);
#pragma unroll
        for (int j = 0; j < 2; ++j) {
#ifdef USE_TDM
            const __bf16* kp = &Klds[buf][(j * 16 + l16) * HD];
#else
            const __bf16* kp = Kp + (size_t)(key0 + j * 16 + l16) * HD;
#endif
#pragma unroll
            for (int kc = 0; kc < 4; ++kc) {
                v16bf bk = cat16(ld8(kp + kc * 32 + half * 8), ld8(kp + kc * 32 + 16 + half * 8));
                sacc[j] = wmma_bf16(qf[kc], bk, sacc[j]);
            }
        }
        // ---- online softmax per row (row m = half*8+r over 16 lanes) ----
#pragma unroll
        for (int r = 0; r < 8; ++r) {
            float s0v = sacc[0][r] * scale;
            float s1v = sacc[1][r] * scale;
            float mx = fmaxf(s0v, s1v);
#pragma unroll
            for (int off = 1; off < 16; off <<= 1)
                mx = fmaxf(mx, __shfl_xor(mx, off, 32));
            const float mnew = fmaxf(rm[r], mx);
            const float corr = __expf(rm[r] - mnew);
            rm[r] = mnew;
            const float p0 = __expf(s0v - mnew);
            const float p1 = __expf(s1v - mnew);
            sacc[0][r] = p0; sacc[1][r] = p1;
            float rs = p0 + p1;
#pragma unroll
            for (int off = 1; off < 16; off <<= 1)
                rs += __shfl_xor(rs, off, 32);
            rl[r] = rl[r] * corr + rs;
#pragma unroll
            for (int t = 0; t < 8; ++t) o[t][r] *= corr;
        }
        // ---- P: C-layout -> A-fragment via LDS ----
        __syncthreads();
#pragma unroll
        for (int j = 0; j < 2; ++j)
#pragma unroll
            for (int r = 0; r < 8; ++r)
                Plds[(half * 8 + r) * 32 + j * 16 + l16] = (__bf16)sacc[j][r];
        __syncthreads();
        const __bf16* pp = Plds + l16 * 32;
        v16bf pa = cat16(*(const v8bf*)(pp + half * 8),
                         *(const v8bf*)(pp + 16 + half * 8));
        // ---- PV ----
#pragma unroll
        for (int t = 0; t < 8; ++t) {
#ifdef USE_TDM
            const __bf16* vp = &Vlds[buf][(t * 16 + l16) * 32];
#else
            const __bf16* vp = Vp + (size_t)(t * 16 + l16) * SS + key0;
#endif
            v16bf bv = cat16(ld8(vp + half * 8), ld8(vp + 16 + half * 8));
            o[t] = wmma_bf16(pa, bv, o[t]);
        }
    }

    // ---- finalize ----
#pragma unroll
    for (int r = 0; r < 8; ++r) {
        const int s = q0 + half * 8 + r;
        const float inv = 1.0f / rl[r];
#pragma unroll
        for (int t = 0; t < 8; ++t)
            Ab[((size_t)b * SS + s) * (NH * HD) + h * HD + t * 16 + l16] =
                (__bf16)(o[t][r] * inv);
    }
}

// =====================================================================
// Kernel 3: output projection  out[f32] = attn[bf16] @ Wo (bf16, transposed)
// =====================================================================
__global__ __launch_bounds__(32)
void oproj_kernel(const __bf16* __restrict__ Ab,
                  const __bf16* __restrict__ Wot,   // [HID][NH*HD]
                  float* __restrict__ Out)
{
    const int lane = threadIdx.x & 31;
    const int half = lane >> 4;
    const int l16  = lane & 15;

    const int token0  = blockIdx.x * 16;
    const int colbase = blockIdx.y * 128;

    v8f acc[8];
#pragma unroll
    for (int t = 0; t < 8; ++t) acc[t] = (v8f)(0.0f);

    const __bf16* ap = Ab + (size_t)(token0 + l16) * (NH * HD);

    for (int k0 = 0; k0 < NH * HD; k0 += 32) {
        v16bf a = cat16(ld8(ap + k0 + half * 8),
                        ld8(ap + k0 + 16 + half * 8));
#pragma unroll
        for (int t = 0; t < 8; ++t) {
            const __bf16* wp = Wot + (size_t)(colbase + t * 16 + l16) * (NH * HD)
                                   + k0;
            v16bf bfr = cat16(ld8(wp + half * 8), ld8(wp + 16 + half * 8));
            acc[t] = wmma_bf16(a, bfr, acc[t]);
        }
    }
#pragma unroll
    for (int r = 0; r < 8; ++r) {
        const int row = token0 + half * 8 + r;
#pragma unroll
        for (int t = 0; t < 8; ++t)
            Out[(size_t)row * HID + colbase + t * 16 + l16] = acc[t][r];
    }
}

// =====================================================================
extern "C" void kernel_launch(void* const* d_in, const int* in_sizes, int n_in,
                              void* d_out, int out_size, void* d_ws, size_t ws_size,
                              hipStream_t stream) {
    const float* H    = (const float*)d_in[0];
    const float* cosT = (const float*)d_in[1];
    const float* sinT = (const float*)d_in[2];
    const float* Wq   = (const float*)d_in[3];
    const float* Wk   = (const float*)d_in[4];
    const float* Wv   = (const float*)d_in[5];
    const float* Wo   = (const float*)d_in[6];
    float* Out        = (float*)d_out;

    // ---- workspace partition (all bf16) ----
    __bf16* Hbf = (__bf16*)d_ws;                              // [B*S][HID]
    __bf16* Wqt = Hbf + (size_t)BB * SS * HID;                // [2048][2048]
    __bf16* Wkt = Wqt + (size_t)(NH  * HD) * HID;             // [512][2048]
    __bf16* Wvt = Wkt + (size_t)(NKV * HD) * HID;             // [512][2048]
    __bf16* Wot = Wvt + (size_t)(NKV * HD) * HID;             // [2048][2048]
    __bf16* Qb  = Wot + (size_t)HID * (NH * HD);              // [B][NH][S][HD]
    __bf16* Kb  = Qb  + (size_t)BB * NH  * SS * HD;           // [B][NKV][S][HD]
    __bf16* Vt  = Kb  + (size_t)BB * NKV * SS * HD;           // [B][NKV][HD][S]
    __bf16* Ab  = Vt  + (size_t)BB * NKV * SS * HD;           // [B*S][NH*HD]

    // ---- prep ----
    {
        int n = BB * SS * HID;
        hcvt_kernel<<<(n + 255) / 256, 256, 0, stream>>>(H, Hbf, n);
    }
    dim3 tblk(32, 8, 1);
    wtrans_kernel<<<dim3((NH  * HD) / 32, HID / 32), tblk, 0, stream>>>(Wq, Wqt, HID, NH  * HD);
    wtrans_kernel<<<dim3((NKV * HD) / 32, HID / 32), tblk, 0, stream>>>(Wk, Wkt, HID, NKV * HD);
    wtrans_kernel<<<dim3((NKV * HD) / 32, HID / 32), tblk, 0, stream>>>(Wv, Wvt, HID, NKV * HD);
    wtrans_kernel<<<dim3(HID / 32, (NH * HD) / 32), tblk, 0, stream>>>(Wo, Wot, NH * HD, HID);

    dim3 blk(32, 1, 1);

    // 1) QKV + RoPE
    qkv_rope_kernel<<<dim3((BB * SS) / 16, NH + 2 * NKV, 1), blk, 0, stream>>>(
        Hbf, cosT, sinT, Wqt, Wkt, Wvt, Qb, Kb, Vt);

    // 2) flash attention (TDM double-buffered)
    attn_kernel<<<dim3(SS / 16, NH, BB), blk, 0, stream>>>(Qb, Kb, Vt, Ab);

    // 3) output projection
    oproj_kernel<<<dim3((BB * SS) / 16, HID / 128, 1), blk, 0, stream>>>(Ab, Wot, Out);
}
